// Attention_44487271252362
// MI455X (gfx1250) — matrix operand order repaired and checked by reading, not compiled, not run on hardware
//
#include <hip/hip_runtime.h>

typedef unsigned short u16;
typedef __attribute__((ext_vector_type(16))) __bf16 v16bf;
typedef __attribute__((ext_vector_type(8)))  float  v8f;

union FragBF {
    v16bf v;
    uint4 q[2];
    u16   u[16];
};

__device__ __forceinline__ u16 f2bf(float f) {
    unsigned u = __float_as_uint(f);
    unsigned r = u + 0x7FFFu + ((u >> 16) & 1u);
    return (u16)(r >> 16);
}
__device__ __forceinline__ float bf2f(u16 h) {
    return __uint_as_float(((unsigned)h) << 16);
}
__device__ __forceinline__ v8f vzero8() {
    v8f z;
#pragma unroll
    for (int i = 0; i < 8; i++) z[i] = 0.0f;
    return z;
}

// ---------------------------------------------------------------------------
// CDNA5 async global->LDS staging (ASYNCcnt path), with safe fallback.
// Builtin signature (from compiler diagnostic): param0 is a non-const
// global (AS1) pointer to int __attribute__((vector_size(16))).
// ---------------------------------------------------------------------------
typedef int v4i_vs __attribute__((vector_size(16)));
typedef __attribute__((address_space(1))) v4i_vs* glob_v4i;
typedef __attribute__((address_space(3))) v4i_vs* lds_v4i;

#if __has_builtin(__builtin_amdgcn_global_load_async_to_lds_b128)
#define HAVE_ASYNC_LDS 1
__device__ __forceinline__ void async_wait0() {
#if __has_builtin(__builtin_amdgcn_s_wait_asynccnt)
    __builtin_amdgcn_s_wait_asynccnt(0);
#else
    asm volatile("s_wait_asynccnt 0x0" ::: "memory");
#endif
}
#endif

// Copy N128 * 16 bytes from global to LDS (per calling thread).
template <int N128>
__device__ __forceinline__ void lds_copy(const u16* __restrict__ src, u16* dst) {
#ifdef HAVE_ASYNC_LDS
#pragma unroll
    for (int i = 0; i < N128; i++)
        __builtin_amdgcn_global_load_async_to_lds_b128(
            (glob_v4i)(src + i * 8), (lds_v4i)(dst + i * 8), 0, 0);
#else
#pragma unroll
    for (int i = 0; i < N128; i++)
        ((uint4*)dst)[i] = ((const uint4*)src)[i];
#endif
}
__device__ __forceinline__ void lds_copy_fence() {
#ifdef HAVE_ASYNC_LDS
    async_wait0();
#endif
}

// Problem constants (from setup_inputs)
#define BB 2
#define TT 2048
#define DD 2048
#define NH 16
#define KH 4
#define HH 128
#define SS 2048
#define ROPE_DIM 64
#define SLIDING_WINDOW 1024
#define ATT_SCALE 0.08838834764831843f   // 128^-0.5
#define NEG_BIG  (-3.0e38f)

// ---------------------------------------------------------------------------
// f32 -> bf16 conversion
// ---------------------------------------------------------------------------
__global__ __launch_bounds__(256) void f32_to_bf16_kernel(const float* __restrict__ in,
                                                          u16* __restrict__ out, size_t n) {
    size_t i = (size_t)blockIdx.x * blockDim.x + threadIdx.x;
    size_t stride = (size_t)gridDim.x * blockDim.x;
    for (; i < n; i += stride) out[i] = f2bf(in[i]);
}

// ---------------------------------------------------------------------------
// Positions / segment start prep
// ---------------------------------------------------------------------------
__global__ void prep_kernel(const int* __restrict__ seg, const int* __restrict__ start_ind,
                            const int* __restrict__ curp, int* __restrict__ pos,
                            int* __restrict__ startv) {
    int b = blockIdx.x;
    __shared__ int fn_s;
    if (threadIdx.x == 0) {
        int fn = -1;
        for (int t = 0; t < TT; t++) {
            if (seg[b * TT + t] != 0) { fn = t; break; }
        }
        fn_s = fn;
        int lp = (fn < 0) ? TT : fn;
        int st = start_ind[b];
        startv[b] = (st < 0) ? lp : st;
    }
    __syncthreads();
    int fn = fn_s;
    int argm = (fn < 0) ? 0 : fn;
    int cur = curp[0];
    for (int t = threadIdx.x; t < TT; t += blockDim.x) {
        pos[b * TT + t] = (seg[b * TT + t] != 0) ? (t - argm + cur) : ((1 << 30) + cur);
    }
}

// ---------------------------------------------------------------------------
// Generic bf16 GEMM: C[MxN] = A[MxK] * B[KxN], A/B row-major bf16.
// 256 threads = 8 waves; block tile 128x128; K-step 32; WMMA bf16 16x16x32.
// OUT_BF16 templated so the epilogue is branch-free.
// ---------------------------------------------------------------------------
template <int OUT_BF16>
__global__ __launch_bounds__(256) void gemm_bf16_kernel(const u16* __restrict__ A,
                                                        const u16* __restrict__ Bm,
                                                        void* __restrict__ Cout,
                                                        int M, int N, int Kd) {
    __shared__ __align__(16) u16 lA[128 * 32];   // [row][k]
    __shared__ __align__(16) u16 lB[128 * 32];   // [col][k]  (transposed)

    int tid = threadIdx.x;
    int wave = tid >> 5, lane = tid & 31;
    int ln = lane & 15, lh = lane >> 4;
    int m0 = blockIdx.y * 128;
    int n0 = blockIdx.x * 128;

    v8f acc[8];
#pragma unroll
    for (int j = 0; j < 8; j++) acc[j] = vzero8();

    int arow = tid >> 1, ahalf = tid & 1;
    int bk = tid >> 3, bcg = tid & 7;

    for (int kb = 0; kb < Kd; kb += 32) {
        {   // A tile: 128 rows x 32 bf16, each thread = half row (async to LDS)
            const u16* src = A + (size_t)(m0 + arow) * Kd + kb + ahalf * 16;
            lds_copy<2>(src, &lA[arow * 32 + ahalf * 16]);
            if (kb + 32 < Kd) __builtin_prefetch(src + 32, 0, 1);
        }
        {   // B tile 32x128 -> store transposed [col][k]
            const u16* src = Bm + (size_t)(kb + bk) * N + n0 + bcg * 16;
            union { uint4 q; u16 u[8]; } t0, t1;
            t0.q = *(const uint4*)src;
            t1.q = *(const uint4*)(src + 8);
            if (kb + 32 < Kd) __builtin_prefetch(src + (size_t)32 * N, 0, 1);
#pragma unroll
            for (int c = 0; c < 8; c++) lB[(bcg * 16 + c) * 32 + bk] = t0.u[c];
#pragma unroll
            for (int c = 0; c < 8; c++) lB[(bcg * 16 + 8 + c) * 32 + bk] = t1.u[c];
        }
        lds_copy_fence();
        __syncthreads();

        FragBF fa;
        const u16* ap = &lA[(wave * 16 + ln) * 32 + lh * 16];
        fa.q[0] = *(const uint4*)ap;          // K = {0..7} or {16..23}
        fa.q[1] = *(const uint4*)(ap + 8);    // K = {8..15} or {24..31}
#pragma unroll
        for (int j = 0; j < 8; j++) {
            FragBF fb;
            const u16* bp = &lB[(j * 16 + ln) * 32 + lh * 16];
            fb.q[0] = *(const uint4*)bp;
            fb.q[1] = *(const uint4*)(bp + 8);
            acc[j] = __builtin_amdgcn_wmma_f32_16x16x32_bf16(
                false, fa.v, false, fb.v, (short)0, acc[j], false, false);
        }
        __syncthreads();
    }

#pragma unroll
    for (int j = 0; j < 8; j++) {
#pragma unroll
        for (int r = 0; r < 8; r++) {
            int rr = lh ? (r + 8) : r;
            size_t idx = (size_t)(m0 + wave * 16 + rr) * N + n0 + j * 16 + ln;
            float v = acc[j][r];
            if (OUT_BF16) ((u16*)Cout)[idx] = f2bf(v);
            else          ((float*)Cout)[idx] = v;
        }
    }
}

// ---------------------------------------------------------------------------
// RoPE on q in place (bf16), one thread per (row, head)
// ---------------------------------------------------------------------------
__global__ __launch_bounds__(256) void rope_q_kernel(u16* __restrict__ q,
                                                     const int* __restrict__ pos) {
    int idx = blockIdx.x * blockDim.x + threadIdx.x;       // (b*T+t)*NH + n
    if (idx >= BB * TT * NH) return;
    int n = idx % NH, row = idx / NH;
    float fp = (float)pos[row];
    size_t base = (size_t)row * (NH * HH) + (size_t)n * HH;
#pragma unroll 4
    for (int j = 0; j < ROPE_DIM / 2; j++) {
        float freq = __powf(10000.0f, -(float)(2 * j) / (float)ROPE_DIM);
        float a = fp * freq;
        float sn = __sinf(a), cs = __cosf(a);
        float x1 = bf2f(q[base + j]);
        float x2 = bf2f(q[base + ROPE_DIM / 2 + j]);
        q[base + j]                = f2bf(x1 * cs - x2 * sn);
        q[base + ROPE_DIM / 2 + j] = f2bf(x2 * cs + x1 * sn);
    }
}

// ---------------------------------------------------------------------------
// RoPE on k + write k,v into bf16 caches at s = cur + t
// ---------------------------------------------------------------------------
__global__ __launch_bounds__(256) void ropekv_kernel(const u16* __restrict__ ktmp,
                                                     const u16* __restrict__ vtmp,
                                                     u16* __restrict__ kcb,
                                                     u16* __restrict__ vcb,
                                                     const int* __restrict__ pos,
                                                     const int* __restrict__ curp) {
    int idx = blockIdx.x * blockDim.x + threadIdx.x;       // (b*T+t)*KH + g
    if (idx >= BB * TT * KH) return;
    int g = idx % KH, row = idx / KH;
    int b = row / TT, t = row % TT;
    int cur = curp[0];
    int s = cur + t;
    if (s >= SS) return;
    float fp = (float)pos[row];
    size_t src = (size_t)row * (KH * HH) + (size_t)g * HH;
    size_t dst = ((size_t)b * SS + s) * (KH * HH) + (size_t)g * HH;
#pragma unroll 4
    for (int j = 0; j < ROPE_DIM / 2; j++) {
        float freq = __powf(10000.0f, -(float)(2 * j) / (float)ROPE_DIM);
        float a = fp * freq;
        float sn = __sinf(a), cs = __cosf(a);
        float x1 = bf2f(ktmp[src + j]);
        float x2 = bf2f(ktmp[src + ROPE_DIM / 2 + j]);
        kcb[dst + j]                = f2bf(x1 * cs - x2 * sn);
        kcb[dst + ROPE_DIM / 2 + j] = f2bf(x2 * cs + x1 * sn);
    }
#pragma unroll 8
    for (int h = ROPE_DIM; h < HH; h++) kcb[dst + h] = ktmp[src + h];
#pragma unroll 8
    for (int h = 0; h < HH; h++) vcb[dst + h] = vtmp[src + h];
}

// ---------------------------------------------------------------------------
// Flash attention: grid (T/128, NH, BB), 256 threads = 8 waves.
// Q in registers; K then V chunk in one LDS buffer (async staged);
// P re-layout via per-wave LDS tile.
// ---------------------------------------------------------------------------
__global__ __launch_bounds__(256) void attn_kernel(const u16* __restrict__ qb,
                                                   const u16* __restrict__ kcb,
                                                   const u16* __restrict__ vcb,
                                                   const float* __restrict__ sink_bias,
                                                   const int* __restrict__ seg,
                                                   const int* __restrict__ startv,
                                                   const int* __restrict__ curp,
                                                   u16* __restrict__ attnb) {
    __shared__ __align__(16) u16 lKV[128 * 128];          // K chunk, then V chunk (32KB)
    __shared__ __align__(16) u16 lP[8 * 16 * 128];        // per-wave P tiles (32KB)

    int tile = blockIdx.x, n = blockIdx.y, b = blockIdx.z;
    int g = n / (NH / KH);
    int tid = threadIdx.x, wave = tid >> 5, lane = tid & 31;
    int ln = lane & 15, lh = lane >> 4;
    int cur = curp[0];
    int start = startv[b];

    // Q fragments for this wave's 16 rows (A-fragment layout, 4 K-steps of 32)
    int t_frag = tile * 128 + wave * 16 + ln;
    size_t qrow = (size_t)(b * TT + t_frag) * (NH * HH) + (size_t)n * HH;
    FragBF qf[4];
#pragma unroll
    for (int kk = 0; kk < 4; kk++) {
        const u16* qp_ = qb + qrow + kk * 32 + lh * 16;
        qf[kk].q[0] = *(const uint4*)qp_;
        qf[kk].q[1] = *(const uint4*)(qp_ + 8);
    }

    // per-row softmax state: rows r (lanes 0-15) / r+8 (lanes 16-31)
    float snk = sink_bias[n];
    float mr[8], ls[8];
    int qp[8], qs[8];
    v8f oacc[8];
#pragma unroll
    for (int r = 0; r < 8; r++) {
        mr[r] = snk;          // sink participates in the max
        ls[r] = 1.0f;         // exp(sink - m) = 1 at init
        int te = tile * 128 + wave * 16 + (lh ? (r + 8) : r);
        qp[r] = cur + te - start;
        qs[r] = seg[b * TT + te];
        oacc[r] = vzero8();
    }

    int s_begin = cur + tile * 128 - (SLIDING_WINDOW - 1);
    if (s_begin < 0) s_begin = 0;
    int s_end = cur + tile * 128 + 127;
    if (s_end > SS - 1) s_end = SS - 1;

    int sl = tid >> 1, hh = (tid & 1) * 64;     // cooperative-copy assignment
    int c0 = s_begin >> 7, c1 = s_end >> 7;

    for (int c = c0; c <= c1; ++c) {
        int s0 = c << 7;
        size_t kvrow = ((size_t)b * SS + (s0 + sl)) * (KH * HH) + (size_t)g * HH + hh;

        // stage K chunk, row-major [s_local][h] (async to LDS)
        lds_copy<8>(kcb + kvrow, &lKV[sl * 128 + hh]);
        if (c < c1) __builtin_prefetch(kcb + kvrow + (size_t)128 * (KH * HH), 0, 1);
        lds_copy_fence();
        __syncthreads();

        // logits: 8 s-subtiles x (4 WMMA over H)
        v8f sacc[8];
#pragma unroll
        for (int j = 0; j < 8; j++) {
            sacc[j] = vzero8();
#pragma unroll
            for (int kk = 0; kk < 4; kk++) {
                FragBF fb;
                const u16* bp = &lKV[(j * 16 + ln) * 128 + kk * 32 + lh * 16];
                fb.q[0] = *(const uint4*)bp;
                fb.q[1] = *(const uint4*)(bp + 8);
                sacc[j] = __builtin_amdgcn_wmma_f32_16x16x32_bf16(
                    false, qf[kk].v, false, fb.v, (short)0, sacc[j], false, false);
            }
        }

        // scale + mask + chunk row-max
        float cm[8];
#pragma unroll
        for (int r = 0; r < 8; r++) cm[r] = NEG_BIG;
#pragma unroll
        for (int j = 0; j < 8; j++) {
            int s = s0 + j * 16 + ln;
            int kseg = (s >= start && s < cur + TT) ? 1 : 0;
            int kpos = s - start;
#pragma unroll
            for (int r = 0; r < 8; r++) {
                float v = sacc[j][r] * ATT_SCALE;
                bool ok = (kpos <= qp[r]) && (kpos >= qp[r] - (SLIDING_WINDOW - 1)) &&
                          (kseg == qs[r]);
                v = ok ? v : NEG_BIG;
                sacc[j][r] = v;
                cm[r] = fmaxf(cm[r], v);
            }
        }
#pragma unroll
        for (int r = 0; r < 8; r++) {
#pragma unroll
            for (int off = 8; off >= 1; off >>= 1)
                cm[r] = fmaxf(cm[r], __shfl_xor(cm[r], off, 32));
        }

        // online softmax update + emit P (bf16) to per-wave LDS tile
        float rs[8];
#pragma unroll
        for (int r = 0; r < 8; r++) {
            float nm = fmaxf(mr[r], cm[r]);
            float al = __expf(mr[r] - nm);
            mr[r] = nm;
            ls[r] *= al;
            rs[r] = 0.0f;
#pragma unroll
            for (int o = 0; o < 8; o++) oacc[o][r] *= al;
        }
#pragma unroll
        for (int j = 0; j < 8; j++) {
#pragma unroll
            for (int r = 0; r < 8; r++) {
                float p = __expf(sacc[j][r] - mr[r]);
                rs[r] += p;
                lP[wave * 2048 + (lh ? (r + 8) : r) * 128 + j * 16 + ln] = f2bf(p);
            }
        }
#pragma unroll
        for (int r = 0; r < 8; r++) {
#pragma unroll
            for (int off = 8; off >= 1; off >>= 1) rs[r] += __shfl_xor(rs[r], off, 32);
            ls[r] += rs[r];
        }

        __syncthreads();   // everyone done with K before overwriting with V
        lds_copy<8>(vcb + kvrow, &lKV[sl * 128 + hh]);   // stage V chunk
        lds_copy_fence();
        __syncthreads();

        // P A-fragments (4 K-steps over s)
        FragBF pf[4];
#pragma unroll
        for (int kk = 0; kk < 4; kk++) {
            const u16* pp = &lP[wave * 2048 + ln * 128 + kk * 32 + lh * 16];
            pf[kk].q[0] = *(const uint4*)pp;
            pf[kk].q[1] = *(const uint4*)(pp + 8);
        }
        // O += P * V
#pragma unroll
        for (int o = 0; o < 8; o++) {
#pragma unroll
            for (int kk = 0; kk < 4; kk++) {
                FragBF fv;
#pragma unroll
                for (int jj = 0; jj < 16; jj++)
                    fv.u[jj] = lKV[(kk * 32 + lh * 16 + jj) * 128 + o * 16 + ln];
                oacc[o] = __builtin_amdgcn_wmma_f32_16x16x32_bf16(
                    false, pf[kk].v, false, fv.v, (short)0, oacc[o], false, false);
            }
        }
        __syncthreads();   // before next chunk reuses lKV
    }

    // finalize: divide by softmax sum (includes sink term), store bf16
#pragma unroll
    for (int r = 0; r < 8; r++) {
        float inv = 1.0f / ls[r];
        int te = tile * 128 + wave * 16 + (lh ? (r + 8) : r);
        size_t orow = (size_t)(b * TT + te) * (NH * HH) + (size_t)n * HH;
#pragma unroll
        for (int o = 0; o < 8; o++)
            attnb[orow + o * 16 + ln] = f2bf(oacc[o][r] * inv);
    }
}

// ---------------------------------------------------------------------------
// Launch
// ---------------------------------------------------------------------------
extern "C" void kernel_launch(void* const* d_in, const int* in_sizes, int n_in,
                              void* d_out, int out_size, void* d_ws, size_t ws_size,
                              hipStream_t stream) {
    const float* x    = (const float*)d_in[0];
    const float* wq   = (const float*)d_in[1];
    const float* wk   = (const float*)d_in[2];
    const float* wv   = (const float*)d_in[3];
    const float* wo   = (const float*)d_in[4];
    const float* sink = (const float*)d_in[5];
    const int* seg    = (const int*)d_in[8];
    const int* curp   = (const int*)d_in[9];
    const int* st_ind = (const int*)d_in[10];

    char* ws = (char*)d_ws;
    u16* xb    = (u16*)(ws + 0);              // 16,777,216 B
    u16* wqb   = (u16*)(ws + 16777216);       //  8,388,608
    u16* wkb   = (u16*)(ws + 25165824);       //  2,097,152
    u16* wvb   = (u16*)(ws + 27262976);       //  2,097,152
    u16* wob   = (u16*)(ws + 29360128);       //  8,388,608
    u16* qbuf  = (u16*)(ws + 37748736);       // 16,777,216
    u16* ktmp  = (u16*)(ws + 54525952);       //  4,194,304
    u16* vtmp  = (u16*)(ws + 58720256);       //  4,194,304
    u16* kcb   = (u16*)(ws + 62914560);       //  4,194,304
    u16* vcb   = (u16*)(ws + 67108864);       //  4,194,304
    u16* attnb = (u16*)(ws + 71303168);       // 16,777,216
    int* posb  = (int*)(ws + 88080384);       //     16,384
    int* startb= (int*)(ws + 88096768);       //         64

    // 1) convert to bf16
    f32_to_bf16_kernel<<<2048, 256, 0, stream>>>(x,  xb,  (size_t)BB * TT * DD);
    f32_to_bf16_kernel<<<1024, 256, 0, stream>>>(wq, wqb, (size_t)DD * NH * HH);
    f32_to_bf16_kernel<<<512,  256, 0, stream>>>(wk, wkb, (size_t)DD * KH * HH);
    f32_to_bf16_kernel<<<512,  256, 0, stream>>>(wv, wvb, (size_t)DD * KH * HH);
    f32_to_bf16_kernel<<<1024, 256, 0, stream>>>(wo, wob, (size_t)NH * HH * DD);

    // 2) positions + segment starts
    prep_kernel<<<BB, 256, 0, stream>>>(seg, st_ind, curp, posb, startb);

    // 3) projections (WMMA GEMMs, bf16 out)
    gemm_bf16_kernel<1><<<dim3(NH * HH / 128, BB * TT / 128), 256, 0, stream>>>(
        xb, wqb, qbuf, BB * TT, NH * HH, DD);
    gemm_bf16_kernel<1><<<dim3(KH * HH / 128, BB * TT / 128), 256, 0, stream>>>(
        xb, wkb, ktmp, BB * TT, KH * HH, DD);
    gemm_bf16_kernel<1><<<dim3(KH * HH / 128, BB * TT / 128), 256, 0, stream>>>(
        xb, wvb, vtmp, BB * TT, KH * HH, DD);

    // 4) RoPE + cache fill
    rope_q_kernel<<<(BB * TT * NH + 255) / 256, 256, 0, stream>>>(qbuf, posb);
    ropekv_kernel<<<(BB * TT * KH + 255) / 256, 256, 0, stream>>>(ktmp, vtmp, kcb, vcb,
                                                                  posb, curp);

    // 5) flash attention (WMMA)
    attn_kernel<<<dim3(TT / 128, NH, BB), 256, 0, stream>>>(qbuf, kcb, vcb, sink, seg,
                                                            startb, curp, attnb);

    // 6) output projection -> f32 d_out
    gemm_bf16_kernel<0><<<dim3(DD / 128, BB * TT / 128), 256, 0, stream>>>(
        attnb, wob, d_out, BB * TT, DD, NH * HH);
}
